// cylinder_fea_19421842113124
// MI455X (gfx1250) — matrix operand order repaired and checked by reading, not compiled, not run on hardware
//
#include <hip/hip_runtime.h>

typedef __attribute__((ext_vector_type(16))) __bf16 v16bf;
typedef __attribute__((ext_vector_type(8)))  float  v8f;

#define NSEG   131072
#define NPTS   800000
#define BN_EPS 1e-5f

// ---------- helpers ----------
__device__ __forceinline__ __bf16 f2bf(float f) {
    unsigned u = __float_as_uint(f);
    unsigned r = (u + 0x7FFFu + ((u >> 16) & 1u)) >> 16;
    unsigned short s = (unsigned short)r;
    __bf16 out;
    __builtin_memcpy(&out, &s, 2);
    return out;
}
// monotone float -> u32 encoding so integer atomicMax == float max
__device__ __forceinline__ unsigned encf(float f) {
    unsigned u = __float_as_uint(f);
    return (u & 0x80000000u) ? ~u : (u | 0x80000000u);
}
__device__ __forceinline__ float decf(unsigned e) {
    unsigned u = (e & 0x80000000u) ? (e & 0x7FFFFFFFu) : ~e;
    return __uint_as_float(u);
}

// ---------- phase 0: init ----------
__global__ void k_init_pool(unsigned* pool, unsigned* cnt) {
    int t = blockIdx.x * 256 + threadIdx.x;          // covers NSEG*9 + NSEG words
    if (t < NSEG * 9) pool[t] = 0u;                  // enc(-inf) > 0, so 0 is identity
    if (t < NSEG) cnt[t] = 0u;
}

__global__ void k_prefill(long long* unq, __bf16* X0) {
    int i = blockIdx.x * 256 + threadIdx.x;          // one thread per voxel row
    if (i >= NSEG) return;
    long long* u = unq + 4 * (size_t)i;
    u[0] = -1ll; u[1] = -1ll; u[2] = -1ll; u[3] = -1ll;
    uint4 z = {0u, 0u, 0u, 0u};
    uint4* x = (uint4*)(X0 + 32 * (size_t)i);        // X0 row stride 32 bf16 = 64B
    x[0] = z; x[1] = z; x[2] = z; x[3] = z;
}

// ---------- phase 1: scatter-max ----------
__global__ void k_scatter(const float* __restrict__ fea, const int* __restrict__ ind,
                          unsigned* pool, unsigned* cnt) {
    int t = blockIdx.x * 256 + threadIdx.x;
    if (t >= NPTS) return;
    int b  = t / 200000;
    int ix = ind[3 * (size_t)t + 0];
    int iy = ind[3 * (size_t)t + 1];
    int iz = ind[3 * (size_t)t + 2];
    unsigned key = (((unsigned)(b * 32 + ix) * 32u + (unsigned)iy) * 32u + (unsigned)iz);
#pragma unroll
    for (int j = 0; j < 9; ++j)
        atomicMax(&pool[(size_t)key * 9 + j], encf(fea[(size_t)t * 9 + j]));
    atomicAdd(&cnt[key], 1u);
}

// ---------- phase 2: exclusive scan of occupancy (ranks in sorted-key order) ----------
__global__ void k_scan1(const unsigned* __restrict__ cnt, unsigned* scanInc, unsigned* bsum) {
    __shared__ unsigned s[256];
    int lid = threadIdx.x;
    int i = blockIdx.x * 256 + lid;
    unsigned v = (cnt[i] > 0u) ? 1u : 0u;
    s[lid] = v;
    __syncthreads();
    for (int off = 1; off < 256; off <<= 1) {
        unsigned t = (lid >= off) ? s[lid - off] : 0u;
        __syncthreads();
        s[lid] += t;
        __syncthreads();
    }
    scanInc[i] = s[lid];
    if (lid == 255) bsum[blockIdx.x] = s[255];
}

__global__ void k_scan2(const unsigned* __restrict__ bsum, unsigned* boff) {
    __shared__ unsigned s[512];
    int lid = threadIdx.x;
    unsigned v = bsum[lid];
    s[lid] = v;
    __syncthreads();
    for (int off = 1; off < 512; off <<= 1) {
        unsigned t = (lid >= off) ? s[lid - off] : 0u;
        __syncthreads();
        s[lid] += t;
        __syncthreads();
    }
    boff[lid] = s[lid] - v;    // exclusive
}

// ---------- phase 3: compact + decode + BN0 fold into X0 (bf16, row stride 32) ----------
__global__ void k_compact(const unsigned* __restrict__ pool, const unsigned* __restrict__ cnt,
                          const unsigned* __restrict__ scanInc, const unsigned* __restrict__ boff,
                          const float* __restrict__ g0, const float* __restrict__ v0,
                          long long* unq, __bf16* X0) {
    int i = blockIdx.x * 256 + threadIdx.x;
    if (i >= NSEG) return;
    if (cnt[i] == 0u) return;
    unsigned r = scanInc[i] - 1u + boff[i >> 8];
    long long* u = unq + 4 * (size_t)r;
    u[0] = (long long)(i >> 15);
    u[1] = (long long)((i >> 10) & 31);
    u[2] = (long long)((i >> 5) & 31);
    u[3] = (long long)(i & 31);
    __bf16* x = X0 + 32 * (size_t)r;
#pragma unroll
    for (int j = 0; j < 9; ++j) {
        float s0 = g0[j] * rsqrtf(v0[j] + BN_EPS);   // BN0 scale; BN0 shift folded into bias1
        x[j] = f2bf(decf(pool[(size_t)i * 9 + j]) * s0);
    }
}

// ---------- fold BN scale into bf16 weights, stored in WMMA B-fragment order ----------
// dst layout: [Kpad/32][N/16][lane 0..31][slot 0..15]; lane = (kin>>4)*16 + (n&15),
// slot = kin&15 (slot order == K order within the lane's half).
__global__ void k_foldw(const float* __restrict__ W, const float* __restrict__ g,
                        const float* __restrict__ v, int K, int N, int Kpad, __bf16* dst) {
    int t = blockIdx.x * 256 + threadIdx.x;
    if (t >= Kpad * N) return;
    int kk = t / N, n = t % N;
    float sc = g ? g[n] * rsqrtf(v[n] + BN_EPS) : 1.0f;
    float val = (kk < K) ? W[(size_t)kk * N + n] * sc : 0.0f;
    int kt = kk >> 5, kin = kk & 31;
    int half = kin >> 4, slot = kin & 15;
    int nt = n >> 4, lr = n & 15;
    size_t idx = (((size_t)kt * (N >> 4) + nt) * 32 + (half * 16 + lr)) * 16 + slot;
    dst[idx] = f2bf(val);
}

// ---------- fold BN shift into bias (layer1 also absorbs BN0 shift @ W1) ----------
__global__ void k_foldb(const float* __restrict__ blin, const float* __restrict__ g,
                        const float* __restrict__ bbn, const float* __restrict__ m,
                        const float* __restrict__ v, int N, float* out,
                        const float* __restrict__ g0, const float* __restrict__ b0,
                        const float* __restrict__ m0, const float* __restrict__ v0,
                        const float* __restrict__ W1, int K1) {
    int n = blockIdx.x * 256 + threadIdx.x;
    if (n >= N) return;
    float s = g[n] * rsqrtf(v[n] + BN_EPS);
    float val = blin[n] * s + (bbn[n] - m[n] * s);
    if (W1) {
        for (int kk = 0; kk < K1; ++kk) {
            float s0 = g0[kk] * rsqrtf(v0[kk] + BN_EPS);
            float t0 = b0[kk] - m0[kk] * s0;
            val += t0 * W1[(size_t)kk * N + n] * s;
        }
    }
    out[n] = val;
}

// ---------- bf16 WMMA GEMM: Y = act(X @ W + bias), K % 32 == 0, no guards ----------
// 8 waves/block; each wave owns a 16x16 tile; block tile 128(M) x 16(N).
// A tile staged memory->LDS via GLOBAL_LOAD_ASYNC_TO_LDS_B128 (ASYNCcnt path).
__global__ __launch_bounds__(256) void k_gemm_bf16(
    const __bf16* __restrict__ A, const __bf16* __restrict__ Bw,
    const float* __restrict__ bias, __bf16* __restrict__ Yb,
    float* __restrict__ Yf, int M, int K, int N, int relu) {
    __shared__ __align__(16) __bf16 As[128][40];     // 80B row stride: 16B-aligned, conflict-free

    const int tid  = threadIdx.x;
    const int wv   = tid >> 5;       // wave id 0..7
    const int lane = tid & 31;
    const int lr   = lane & 15;
    const int half = lane >> 4;
    const int m0   = blockIdx.x * 128;
    const int n    = blockIdx.y * 16 + lr;
    const int nTiles = N >> 4;

    const int srow  = tid >> 1;            // staging row 0..127
    const int spart = (tid & 1) * 16;      // staging half-row (16 bf16 = 32B)
    const __bf16* srcBase = A + (size_t)(m0 + srow) * K + spart;
    const unsigned ldsOff = (unsigned)(size_t)(&As[srow][spart]);  // flat addr[31:0] == LDS offset

    v8f acc = {0.f, 0.f, 0.f, 0.f, 0.f, 0.f, 0.f, 0.f};

    for (int k = 0; k < K; k += 32) {
        // async stage A[m0..m0+127][k..k+31] into LDS: 2x 16B per thread, no VGPR roundtrip.
        // INST_OFFSET is added to both the global and the LDS address.
        {
            const __bf16* gsrc = srcBase + k;
            asm volatile(
                "global_load_async_to_lds_b128 %0, %1, off\n\t"
                "global_load_async_to_lds_b128 %0, %1, off offset:16"
                :
                : "v"(ldsOff), "v"(gsrc)
                : "memory");
            if (k + 32 < K)
                __builtin_prefetch((const char*)(srcBase + k + 32), 0, 1);
            asm volatile("s_wait_asynccnt 0x0" ::: "memory");  // this wave's async LDS writes done
        }
        __syncthreads();

        // A fragment: each lane's 8 K-pairs are two contiguous 16B runs -> 2x ds_load_b128
        v16bf a;
        {
            const char* base = (const char*)&As[wv * 16 + lr][0];
            uint4 alo = *(const uint4*)(base + half * 32);        // K: half*16..half*16+7
            uint4 ahi = *(const uint4*)(base + half * 32 + 16);   // K: half*16+8..half*16+15
            __builtin_memcpy(&a, &alo, 16);
            __builtin_memcpy((char*)&a + 16, &ahi, 16);
        }
        // B fragment: pre-swizzled, one contiguous 32B load per lane (coalesced)
        v16bf b = *((const v16bf*)Bw +
                    ((size_t)(k >> 5) * nTiles + blockIdx.y) * 32 + lane);

        acc = __builtin_amdgcn_wmma_f32_16x16x32_bf16(false, a, false, b,
                                                      (short)0, acc, false, false);
        __syncthreads();
    }

    // epilogue: bias + optional ReLU; D layout: VGPR r -> M = half*8 + r, N = lr
    float bn = bias ? bias[n] : 0.0f;
#pragma unroll
    for (int r = 0; r < 8; ++r) {
        int m = m0 + wv * 16 + half * 8 + r;
        float val = acc[r] + bn;
        if (relu) val = fmaxf(val, 0.0f);
        if (Yb) Yb[(size_t)m * N + n] = f2bf(val);
        if (Yf) Yf[(size_t)m * N + n] = val;
    }
}

// ---------- host ----------
extern "C" void kernel_launch(void* const* d_in, const int* in_sizes, int n_in,
                              void* d_out, int out_size, void* d_ws, size_t ws_size,
                              hipStream_t stream) {
    (void)in_sizes; (void)n_in; (void)out_size; (void)ws_size;

    const float* pt_fea = (const float*)d_in[0];
    const int*   xy_ind = (const int*)d_in[1];
    const float* bn0_g = (const float*)d_in[2],  *bn0_b = (const float*)d_in[3];
    const float* bn0_m = (const float*)d_in[4],  *bn0_v = (const float*)d_in[5];
    const float* bn1_g = (const float*)d_in[6],  *bn1_b = (const float*)d_in[7];
    const float* bn1_m = (const float*)d_in[8],  *bn1_v = (const float*)d_in[9];
    const float* bn2_g = (const float*)d_in[10], *bn2_b = (const float*)d_in[11];
    const float* bn2_m = (const float*)d_in[12], *bn2_v = (const float*)d_in[13];
    const float* bn3_g = (const float*)d_in[14], *bn3_b = (const float*)d_in[15];
    const float* bn3_m = (const float*)d_in[16], *bn3_v = (const float*)d_in[17];
    const float* w1 = (const float*)d_in[18], *b1 = (const float*)d_in[19];
    const float* w2 = (const float*)d_in[20], *b2 = (const float*)d_in[21];
    const float* w3 = (const float*)d_in[22], *b3 = (const float*)d_in[23];
    const float* w4 = (const float*)d_in[24], *b4 = (const float*)d_in[25];
    const float* wc = (const float*)d_in[26], *bc = (const float*)d_in[27];

    long long* unq  = (long long*)d_out;                                    // [131072,4] i64
    float*     outf = (float*)((char*)d_out + (size_t)NSEG * 4 * 8);        // [131072,16] f32

    char* ws = (char*)d_ws;
    size_t off = 0;
    auto alloc = [&](size_t bytes) -> char* {
        char* p = ws + off;
        off = (off + bytes + 255) & ~(size_t)255;
        return p;
    };
    unsigned* pool    = (unsigned*)alloc((size_t)NSEG * 9 * 4);
    unsigned* cnt     = (unsigned*)alloc((size_t)NSEG * 4);
    unsigned* scanInc = (unsigned*)alloc((size_t)NSEG * 4);
    unsigned* bsum    = (unsigned*)alloc(512 * 4);
    unsigned* boff    = (unsigned*)alloc(512 * 4);
    __bf16* X0  = (__bf16*)alloc((size_t)NSEG * 32 * 2);    // K padded to 32
    __bf16* X1  = (__bf16*)alloc((size_t)NSEG * 64 * 2);
    __bf16* X2  = (__bf16*)alloc((size_t)NSEG * 128 * 2);
    __bf16* X3  = (__bf16*)alloc((size_t)NSEG * 256 * 2);
    __bf16* X4  = (__bf16*)alloc((size_t)NSEG * 256 * 2);
    __bf16* W1f = (__bf16*)alloc((size_t)32 * 64 * 2);      // fragment-ordered, K padded
    __bf16* W2f = (__bf16*)alloc((size_t)64 * 128 * 2);
    __bf16* W3f = (__bf16*)alloc((size_t)128 * 256 * 2);
    __bf16* W4f = (__bf16*)alloc((size_t)256 * 256 * 2);
    __bf16* Wcf = (__bf16*)alloc((size_t)256 * 16 * 2);
    float* bias1 = (float*)alloc(64 * 4);
    float* bias2 = (float*)alloc(128 * 4);
    float* bias3 = (float*)alloc(256 * 4);

    dim3 blk(256);

    // init
    k_init_pool<<<dim3((NSEG * 9 + 255) / 256), blk, 0, stream>>>(pool, cnt);
    k_prefill<<<dim3(NSEG / 256), blk, 0, stream>>>(unq, X0);

    // scatter-max
    k_scatter<<<dim3((NPTS + 255) / 256), blk, 0, stream>>>(pt_fea, xy_ind, pool, cnt);

    // ranks (sorted-unique positions)
    k_scan1<<<dim3(NSEG / 256), blk, 0, stream>>>(cnt, scanInc, bsum);
    k_scan2<<<dim3(1), dim3(512), 0, stream>>>(bsum, boff);

    // compact + decode + BN0 scale
    k_compact<<<dim3(NSEG / 256), blk, 0, stream>>>(pool, cnt, scanInc, boff,
                                                    bn0_g, bn0_v, unq, X0);

    // fold weights (BN column scales) to bf16, WMMA B-fragment layout, K padded to mult of 32
    k_foldw<<<dim3((32 * 64 + 255) / 256), blk, 0, stream>>>(w1, bn1_g, bn1_v, 9, 64, 32, W1f);
    k_foldw<<<dim3((64 * 128 + 255) / 256), blk, 0, stream>>>(w2, bn2_g, bn2_v, 64, 128, 64, W2f);
    k_foldw<<<dim3((128 * 256 + 255) / 256), blk, 0, stream>>>(w3, bn3_g, bn3_v, 128, 256, 128, W3f);
    k_foldw<<<dim3((256 * 256 + 255) / 256), blk, 0, stream>>>(w4, nullptr, nullptr, 256, 256, 256, W4f);
    k_foldw<<<dim3((256 * 16 + 255) / 256), blk, 0, stream>>>(wc, nullptr, nullptr, 256, 16, 256, Wcf);

    // fold biases (bias1 absorbs BN0 shift through W1*s1)
    k_foldb<<<dim3(1), blk, 0, stream>>>(b1, bn1_g, bn1_b, bn1_m, bn1_v, 64, bias1,
                                         bn0_g, bn0_b, bn0_m, bn0_v, w1, 9);
    k_foldb<<<dim3(1), blk, 0, stream>>>(b2, bn2_g, bn2_b, bn2_m, bn2_v, 128, bias2,
                                         nullptr, nullptr, nullptr, nullptr, nullptr, 0);
    k_foldb<<<dim3(1), blk, 0, stream>>>(b3, bn3_g, bn3_b, bn3_m, bn3_v, 256, bias3,
                                         nullptr, nullptr, nullptr, nullptr, nullptr, 0);

    // MLP: WMMA GEMM chain (M = 131072), all K multiples of 32 -> branch-free inner loops
    k_gemm_bf16<<<dim3(NSEG / 128, 4), blk, 0, stream>>>(X0, W1f, bias1, X1, nullptr, NSEG, 32, 64, 1);
    k_gemm_bf16<<<dim3(NSEG / 128, 8), blk, 0, stream>>>(X1, W2f, bias2, X2, nullptr, NSEG, 64, 128, 1);
    k_gemm_bf16<<<dim3(NSEG / 128, 16), blk, 0, stream>>>(X2, W3f, bias3, X3, nullptr, NSEG, 128, 256, 1);
    k_gemm_bf16<<<dim3(NSEG / 128, 16), blk, 0, stream>>>(X3, W4f, b4, X4, nullptr, NSEG, 256, 256, 0);
    k_gemm_bf16<<<dim3(NSEG / 128, 1), blk, 0, stream>>>(X4, Wcf, bc, nullptr, outf, NSEG, 256, 16, 1);
}
